// SRNN_77103252898313
// MI455X (gfx1250) — matrix operand order repaired and checked by reading, compiled, not run on hardware
//
#include <hip/hip_runtime.h>

// ---------------------------------------------------------------------------
// SRNN for MI455X (gfx1250, wave32, WMMA).
//   Phase A: gi = x @ Wih^T + bih        (big parallel bf16 WMMA GEMM)
//   Phase B: GRU scan (serial over T)    (1 workgroup, WMMA per step)
//   Phase C: C1/C2 = h_prev @ W1_h^T + b (big parallel bf16 WMMA GEMMs)
//   Phase D: stochastic scan (serial)    (1 workgroup, WMMA per step,
//                                         only K=256 z-part done serially)
// ---------------------------------------------------------------------------

#define BB 64
#define TTT 512
#define DXX 256
#define DZZ 256
#define DHH 512
#define DMLP 512
#define G3H 1536
#define ROWS (BB * TTT)  // 32768

typedef __attribute__((ext_vector_type(16))) __bf16 v16bf;
typedef __attribute__((ext_vector_type(8))) float v8f;

struct alignas(16) Q16 { unsigned u[4]; };
union FragA { v16bf v; Q16 q[2]; };

__device__ __forceinline__ __bf16 f2bf(float f) {
  unsigned u = __builtin_bit_cast(unsigned, f);
  unsigned r = (u + 0x7FFFu + ((u >> 16) & 1u)) >> 16;
  unsigned short s = (unsigned short)r;
  return __builtin_bit_cast(__bf16, s);
}
__device__ __forceinline__ float bf2f(__bf16 b) {
  unsigned short s = __builtin_bit_cast(unsigned short, b);
  unsigned u = ((unsigned)s) << 16;
  return __builtin_bit_cast(float, u);
}

__device__ __forceinline__ v8f wmma_bf16(v16bf a, v16bf b, v8f c) {
  // D(f32 16x16) = A(16x32 bf16) * B(32x16 bf16) + C
  return __builtin_amdgcn_wmma_f32_16x16x32_bf16(false, a, false, b, (short)0, c,
                                                 false, false);
}

// A fragment: rows m_base..+15, K k_base..+31 of row-major [M][ld] bf16.
// Lane L: m = m_base + (L&15); VGPR0-3 K=klo..klo+7, VGPR4-7 K=klo+16..klo+23,
// klo = (L<16)?0:8  -> two contiguous 16B loads.
__device__ __forceinline__ v16bf load_frag_a(const __bf16* src, int ld,
                                             int m_base, int k_base, int lane) {
  int m = m_base + (lane & 15);
  int klo = (lane < 16) ? 0 : 8;
  const __bf16* p = src + (size_t)m * ld + k_base + klo;
  FragA f;
  f.q[0] = *(const Q16*)p;
  f.q[1] = *(const Q16*)(p + 16);
  return f.v;
}

// B fragment for Out = In * W^T with W row-major [N][ld]: B[k][n] = W[n][k].
// Lane L: n = n_base + (L&15), K=klo..klo+7, klo+16..klo+23, klo = (L<16)?0:8.
__device__ __forceinline__ v16bf load_frag_b(const __bf16* w, int ld,
                                             int n_base, int k_base, int lane) {
  int n = n_base + (lane & 15);
  int koff = (lane < 16) ? 0 : 8;
  const __bf16* p = w + (size_t)n * ld + k_base + koff;
  FragA f;
  f.q[0] = *(const Q16*)p;
  f.q[1] = *(const Q16*)(p + 16);
  return f.v;
}

// ---------------------------------------------------------------------------
// f32 -> bf16 conversion (with optional sub-block extraction)
// ---------------------------------------------------------------------------
__global__ void __launch_bounds__(256) cvt_bf16(const float* __restrict__ src,
                                                __bf16* __restrict__ dst,
                                                int rows, int cols, int stride,
                                                int coloff) {
  int i = blockIdx.x * blockDim.x + threadIdx.x;
  if (i >= rows * cols) return;
  int r = i / cols, c = i - r * cols;
  dst[i] = f2bf(src[(size_t)r * stride + coloff + c]);
}

// ---------------------------------------------------------------------------
// Phase A: gi[t*64+b][n] = bf16( sum_k x[b][t][k]*Wih[n][k] + bih[n] )
// M = 32768 (row r = t*64+b), N = 1536, K = 256. 8 waves/block, 1 tile/wave.
// ---------------------------------------------------------------------------
__global__ void __launch_bounds__(256) gi_gemm(const __bf16* __restrict__ x16,
                                               const __bf16* __restrict__ wih16,
                                               const float* __restrict__ bih,
                                               __bf16* __restrict__ gi16) {
  int gw = blockIdx.x * 8 + (threadIdx.x >> 5);
  int lane = threadIdx.x & 31;
  int nt = gw % 96, mt = gw / 96;  // mt < 2048
  int r = mt * 16 + (lane & 15);
  int bidx = r & 63, tidx = r >> 6;  // row permutation: r = t*64 + b
  const __bf16* arow = x16 + ((size_t)bidx * TTT + tidx) * DXX;
  int klo = (lane < 16) ? 0 : 8;
  v8f acc = {};
  for (int k = 0; k < DXX; k += 32) {
    FragA fa;
    fa.q[0] = *(const Q16*)(arow + k + klo);
    fa.q[1] = *(const Q16*)(arow + k + klo + 16);
    v16bf fb = load_frag_b(wih16, DXX, nt * 16, k, lane);
    acc = wmma_bf16(fa.v, fb, acc);
  }
  int n = nt * 16 + (lane & 15);
  int mo = (lane < 16) ? 0 : 8;
  float bias = bih[n];
  for (int j = 0; j < 8; ++j)
    gi16[(size_t)(mt * 16 + mo + j) * G3H + n] = f2bf(acc[j] + bias);
}

// ---------------------------------------------------------------------------
// Phase C: Cout[r][n] = bias[n] + sum_k hseq[r][k]*W[n][k]   (f32 out)
// M = 32768, N = 512, K = 512.
// ---------------------------------------------------------------------------
__global__ void __launch_bounds__(256) hcontrib_gemm(
    const __bf16* __restrict__ hseq16, const __bf16* __restrict__ w16,
    const float* __restrict__ bias, float* __restrict__ cout) {
  int gw = blockIdx.x * 8 + (threadIdx.x >> 5);
  int lane = threadIdx.x & 31;
  int nt = gw & 31, mt = gw >> 5;  // mt < 2048
  int n = nt * 16 + (lane & 15);
  int mo = (lane < 16) ? 0 : 8;
  v8f acc;
  float bv = bias[n];
  for (int j = 0; j < 8; ++j) acc[j] = bv;
  for (int k = 0; k < DHH; k += 32) {
    v16bf fa = load_frag_a(hseq16, DHH, mt * 16, k, lane);
    v16bf fb = load_frag_b(w16, DHH, nt * 16, k, lane);
    acc = wmma_bf16(fa, fb, acc);
  }
  for (int j = 0; j < 8; ++j)
    cout[(size_t)(mt * 16 + mo + j) * DMLP + n] = acc[j];
}

// ---------------------------------------------------------------------------
// Phase B: serial GRU scan. One workgroup, 32 waves. h kept bf16 in LDS.
// Per step: gh = h @ Whh^T (4x96 tiles, 12/wave, K=512), then gates.
// ---------------------------------------------------------------------------
__global__ void __launch_bounds__(1024) gru_scan(
    const __bf16* __restrict__ gi16, const __bf16* __restrict__ whh16,
    const float* __restrict__ bhh, const float* __restrict__ h0,
    __bf16* __restrict__ hseq16, float* __restrict__ ghbuf) {
  __shared__ __bf16 s_h[BB * DHH];  // 64 KB
  int tid = threadIdx.x, lane = tid & 31, wave = tid >> 5;
  for (int i = tid; i < BB * DHH; i += 1024) s_h[i] = f2bf(h0[i & (DHH - 1)]);
  __syncthreads();
  for (int t = 0; t < TTT; ++t) {
    // record h_prev[t] (h at step input) for phase C/D
    for (int i = tid; i < BB * DHH; i += 1024)
      hseq16[(size_t)t * BB * DHH + i] = s_h[i];
    // gh = h @ Whh^T  : 384 tiles of 16x16
    for (int tt = wave; tt < 384; tt += 32) {
      int mt = tt / 96, nt = tt % 96;
      v8f acc = {};
      for (int k = 0; k < DHH; k += 32) {
        v16bf fa = load_frag_a(s_h, DHH, mt * 16, k, lane);
        v16bf fb = load_frag_b(whh16, DHH, nt * 16, k, lane);
        acc = wmma_bf16(fa, fb, acc);
      }
      int n = nt * 16 + (lane & 15);
      int mo = (lane < 16) ? 0 : 8;
      for (int j = 0; j < 8; ++j)
        ghbuf[(size_t)(mt * 16 + mo + j) * G3H + n] = acc[j];
    }
    __threadfence();
    __syncthreads();
    // gates (torch order r,z,n), update h in place
    for (int i = tid; i < BB * DHH; i += 1024) {
      int b = i >> 9, c = i & (DHH - 1);
      const __bf16* gir = gi16 + ((size_t)t * BB + b) * G3H;
      const float* ghr = ghbuf + (size_t)b * G3H;
      float g_r = bf2f(gir[c]) + ghr[c] + bhh[c];
      float g_z = bf2f(gir[c + 512]) + ghr[c + 512] + bhh[c + 512];
      float gin = bf2f(gir[c + 1024]);
      float ghn = ghr[c + 1024] + bhh[c + 1024];
      float r = 1.f / (1.f + __expf(-g_r));
      float z = 1.f / (1.f + __expf(-g_z));
      float nn = tanhf(gin + r * ghn);
      float hprev = bf2f(s_h[i]);
      s_h[i] = f2bf((1.f - z) * nn + z * hprev);
    }
    __syncthreads();
  }
}

// ---------------------------------------------------------------------------
// Phase D: serial stochastic scan. One workgroup, 32 waves.
// Per step only the z-dependent K=256 GEMMs are serial; h contributions
// (C1/C2, incl. b1 bias) were precomputed and seed the WMMA accumulators.
// ---------------------------------------------------------------------------
__global__ void __launch_bounds__(1024) stoch_scan(
    const __bf16* __restrict__ dzW1z, const __bf16* __restrict__ dzWloc,
    const float* __restrict__ dz_bloc, const __bf16* __restrict__ dzWsc,
    const float* __restrict__ dz_bsc, const __bf16* __restrict__ dxW1z,
    const __bf16* __restrict__ dxWloc, const float* __restrict__ dx_bloc,
    const __bf16* __restrict__ dxWsc, const float* __restrict__ dx_bsc,
    const float* __restrict__ C1, const float* __restrict__ C2,
    const float* __restrict__ eps, const float* __restrict__ z0,
    __bf16* __restrict__ z16, float* __restrict__ zloc,
    float* __restrict__ zsc, float* __restrict__ out) {
  __shared__ __bf16 s_h[BB * DMLP];  // 64 KB, bf16 MLP hidden
  int tid = threadIdx.x, lane = tid & 31, wave = tid >> 5;
  for (int i = tid; i < BB * DZZ; i += 1024) z16[i] = f2bf(z0[i & (DZZ - 1)]);
  __threadfence();
  __syncthreads();
  for (int t = 0; t < TTT; ++t) {
    const float* C1t = C1 + (size_t)t * BB * DMLP;
    const float* C2t = C2 + (size_t)t * BB * DMLP;
    // hz = relu(C1[t] + z_prev @ dzW1z^T)  : 4x32 tiles, K=256
    for (int tt = wave; tt < 128; tt += 32) {
      int mt = tt >> 5, nt = tt & 31;
      int n = nt * 16 + (lane & 15);
      int mo = (lane < 16) ? 0 : 8;
      v8f acc;
      for (int j = 0; j < 8; ++j) acc[j] = C1t[(size_t)(mt * 16 + mo + j) * DMLP + n];
      for (int k = 0; k < DZZ; k += 32) {
        v16bf fa = load_frag_a(z16, DZZ, mt * 16, k, lane);
        v16bf fb = load_frag_b(dzW1z, DZZ, nt * 16, k, lane);
        acc = wmma_bf16(fa, fb, acc);
      }
      for (int j = 0; j < 8; ++j) {
        float v = acc[j];
        s_h[(mt * 16 + mo + j) * DMLP + n] = f2bf(v > 0.f ? v : 0.f);
      }
    }
    __syncthreads();
    // z_loc / z_scale_pre : 2 x (4x16 tiles), K=512 from LDS
    for (int tt = wave; tt < 128; tt += 32) {
      bool isSc = tt >= 64;
      int id = tt & 63, mt = id >> 4, nt = id & 15;
      const __bf16* W = isSc ? dzWsc : dzWloc;
      const float* bias = isSc ? dz_bsc : dz_bloc;
      int n = nt * 16 + (lane & 15);
      int mo = (lane < 16) ? 0 : 8;
      v8f acc;
      float bv = bias[n];
      for (int j = 0; j < 8; ++j) acc[j] = bv;
      for (int k = 0; k < DMLP; k += 32) {
        v16bf fa = load_frag_a(s_h, DMLP, mt * 16, k, lane);
        v16bf fb = load_frag_b(W, DMLP, nt * 16, k, lane);
        acc = wmma_bf16(fa, fb, acc);
      }
      float* dst = isSc ? zsc : zloc;
      for (int j = 0; j < 8; ++j) dst[(mt * 16 + mo + j) * DZZ + n] = acc[j];
    }
    __threadfence();
    __syncthreads();
    // z = z_loc + softplus(z_scale_pre) * eps  -> out, z16
    for (int i = tid; i < BB * DZZ; i += 1024) {
      int b = i >> 8, c = i & (DZZ - 1);
      float sp = zsc[i];
      float scale = sp > 20.f ? sp : log1pf(__expf(sp));
      float z = zloc[i] + scale * eps[((size_t)b * TTT + t) * DZZ + c];
      out[((size_t)b * TTT + t) * 768 + c] = z;
      z16[i] = f2bf(z);
    }
    __threadfence();
    __syncthreads();
    // hx = relu(C2[t] + z @ dxW1z^T)
    for (int tt = wave; tt < 128; tt += 32) {
      int mt = tt >> 5, nt = tt & 31;
      int n = nt * 16 + (lane & 15);
      int mo = (lane < 16) ? 0 : 8;
      v8f acc;
      for (int j = 0; j < 8; ++j) acc[j] = C2t[(size_t)(mt * 16 + mo + j) * DMLP + n];
      for (int k = 0; k < DZZ; k += 32) {
        v16bf fa = load_frag_a(z16, DZZ, mt * 16, k, lane);
        v16bf fb = load_frag_b(dxW1z, DZZ, nt * 16, k, lane);
        acc = wmma_bf16(fa, fb, acc);
      }
      for (int j = 0; j < 8; ++j) {
        float v = acc[j];
        s_h[(mt * 16 + mo + j) * DMLP + n] = f2bf(v > 0.f ? v : 0.f);
      }
    }
    __syncthreads();
    // x_loc / x_scale -> out directly
    for (int tt = wave; tt < 128; tt += 32) {
      bool isSc = tt >= 64;
      int id = tt & 63, mt = id >> 4, nt = id & 15;
      const __bf16* W = isSc ? dxWsc : dxWloc;
      const float* bias = isSc ? dx_bsc : dx_bloc;
      int colbase = isSc ? 512 : 256;
      int n = nt * 16 + (lane & 15);
      int mo = (lane < 16) ? 0 : 8;
      v8f acc;
      float bv = bias[n];
      for (int j = 0; j < 8; ++j) acc[j] = bv;
      for (int k = 0; k < DMLP; k += 32) {
        v16bf fa = load_frag_a(s_h, DMLP, mt * 16, k, lane);
        v16bf fb = load_frag_b(W, DMLP, nt * 16, k, lane);
        acc = wmma_bf16(fa, fb, acc);
      }
      for (int j = 0; j < 8; ++j) {
        int b = mt * 16 + mo + j;
        float v = acc[j];
        if (isSc) v = v > 20.f ? v : log1pf(__expf(v));
        out[((size_t)b * TTT + t) * 768 + colbase + n] = v;
      }
    }
    __syncthreads();
  }
}

// ---------------------------------------------------------------------------
extern "C" void kernel_launch(void* const* d_in, const int* in_sizes, int n_in,
                              void* d_out, int out_size, void* d_ws,
                              size_t ws_size, hipStream_t stream) {
  (void)in_sizes; (void)n_in; (void)out_size; (void)ws_size;
  const float* x      = (const float*)d_in[0];
  const float* eps    = (const float*)d_in[1];
  const float* Wih    = (const float*)d_in[2];
  const float* Whh    = (const float*)d_in[3];
  const float* bih    = (const float*)d_in[4];
  const float* bhh    = (const float*)d_in[5];
  const float* h0     = (const float*)d_in[6];
  const float* z0     = (const float*)d_in[7];
  const float* dzW1   = (const float*)d_in[8];
  const float* dzb1   = (const float*)d_in[9];
  const float* dzWlc  = (const float*)d_in[10];
  const float* dzblc  = (const float*)d_in[11];
  const float* dzWscl = (const float*)d_in[12];
  const float* dzbscl = (const float*)d_in[13];
  const float* dxW1   = (const float*)d_in[14];
  const float* dxb1   = (const float*)d_in[15];
  const float* dxWlc  = (const float*)d_in[16];
  const float* dxblc  = (const float*)d_in[17];
  const float* dxWscl = (const float*)d_in[18];
  const float* dxbscl = (const float*)d_in[19];
  float* out = (float*)d_out;

  char* ws = (char*)d_ws;
  size_t off = 0;
  auto alloc = [&](size_t bytes) {
    size_t o = off;
    off = (off + bytes + 255) & ~(size_t)255;
    return o;
  };
  size_t o_x16    = alloc((size_t)ROWS * DXX * 2);       // 16 MB
  size_t o_wih    = alloc((size_t)G3H * DXX * 2);
  size_t o_whh    = alloc((size_t)G3H * DHH * 2);
  size_t o_dzw1z  = alloc((size_t)DMLP * DZZ * 2);
  size_t o_dzw1h  = alloc((size_t)DMLP * DHH * 2);
  size_t o_dzwlc  = alloc((size_t)DZZ * DMLP * 2);
  size_t o_dzwsc  = alloc((size_t)DZZ * DMLP * 2);
  size_t o_dxw1z  = alloc((size_t)DMLP * DZZ * 2);
  size_t o_dxw1h  = alloc((size_t)DMLP * DHH * 2);
  size_t o_dxwlc  = alloc((size_t)DXX * DMLP * 2);
  size_t o_dxwsc  = alloc((size_t)DXX * DMLP * 2);
  size_t o_gi16   = alloc((size_t)ROWS * G3H * 2);       // 96 MB
  size_t o_hseq   = alloc((size_t)ROWS * DHH * 2);       // 32 MB
  size_t o_c1     = alloc((size_t)ROWS * DMLP * 4);      // 64 MB
  size_t o_c2     = alloc((size_t)ROWS * DMLP * 4);      // 64 MB
  size_t o_ghbuf  = alloc((size_t)BB * G3H * 4);
  size_t o_z16    = alloc((size_t)BB * DZZ * 2);
  size_t o_zloc   = alloc((size_t)BB * DZZ * 4);
  size_t o_zsc    = alloc((size_t)BB * DZZ * 4);

  __bf16* x16   = (__bf16*)(ws + o_x16);
  __bf16* wih16 = (__bf16*)(ws + o_wih);
  __bf16* whh16 = (__bf16*)(ws + o_whh);
  __bf16* zw1z  = (__bf16*)(ws + o_dzw1z);
  __bf16* zw1h  = (__bf16*)(ws + o_dzw1h);
  __bf16* zwlc  = (__bf16*)(ws + o_dzwlc);
  __bf16* zwsc  = (__bf16*)(ws + o_dzwsc);
  __bf16* xw1z  = (__bf16*)(ws + o_dxw1z);
  __bf16* xw1h  = (__bf16*)(ws + o_dxw1h);
  __bf16* xwlc  = (__bf16*)(ws + o_dxwlc);
  __bf16* xwsc  = (__bf16*)(ws + o_dxwsc);
  __bf16* gi16  = (__bf16*)(ws + o_gi16);
  __bf16* hseq  = (__bf16*)(ws + o_hseq);
  float*  c1    = (float*)(ws + o_c1);
  float*  c2    = (float*)(ws + o_c2);
  float*  ghbuf = (float*)(ws + o_ghbuf);
  __bf16* z16   = (__bf16*)(ws + o_z16);
  float*  zloc  = (float*)(ws + o_zloc);
  float*  zsc   = (float*)(ws + o_zsc);

  auto cvt = [&](const float* s, __bf16* d, int rows, int cols, int stride,
                 int coloff) {
    int n = rows * cols;
    cvt_bf16<<<(n + 255) / 256, 256, 0, stream>>>(s, d, rows, cols, stride,
                                                  coloff);
  };
  // bf16 conversions (weights + x)
  cvt(x, x16, ROWS, DXX, DXX, 0);
  cvt(Wih, wih16, G3H, DXX, DXX, 0);
  cvt(Whh, whh16, G3H, DHH, DHH, 0);
  cvt(dzW1, zw1z, DMLP, DZZ, DZZ + DHH, 0);      // z-part cols [0,256)
  cvt(dzW1, zw1h, DMLP, DHH, DZZ + DHH, DZZ);    // h-part cols [256,768)
  cvt(dzWlc, zwlc, DZZ, DMLP, DMLP, 0);
  cvt(dzWscl, zwsc, DZZ, DMLP, DMLP, 0);
  cvt(dxW1, xw1z, DMLP, DZZ, DZZ + DHH, 0);
  cvt(dxW1, xw1h, DMLP, DHH, DZZ + DHH, DZZ);
  cvt(dxWlc, xwlc, DXX, DMLP, DMLP, 0);
  cvt(dxWscl, xwsc, DXX, DMLP, DMLP, 0);

  // Phase A: gi (2048 x 96 tiles, 8 waves/block)
  gi_gemm<<<(2048 * 96) / 8, 256, 0, stream>>>(x16, wih16, bih, gi16);

  // Phase B: serial GRU scan (single workgroup)
  gru_scan<<<1, 1024, 0, stream>>>(gi16, whh16, bhh, h0, hseq, ghbuf);

  // Phase C: h-contributions of both MLPs (2048 x 32 tiles each)
  hcontrib_gemm<<<(2048 * 32) / 8, 256, 0, stream>>>(hseq, zw1h, dzb1, c1);
  hcontrib_gemm<<<(2048 * 32) / 8, 256, 0, stream>>>(hseq, xw1h, dxb1, c2);

  // Phase D: serial stochastic scan (single workgroup)
  stoch_scan<<<1, 1024, 0, stream>>>(zw1z, zwlc, dzblc, zwsc, dzbscl, xw1z,
                                     xwlc, dxblc, xwsc, dxbscl, c1, c2, eps,
                                     z0, z16, zloc, zsc, out);
}
